// MultiFactorAttention_22007412424960
// MI455X (gfx1250) — matrix operand order repaired and checked by reading, not compiled, not run on hardware
//
#include <hip/hip_runtime.h>
#include <hip/hip_bf16.h>

// Problem constants (from reference)
#define BB   32
#define HH   168
#define DD   512
#define HIST 336
#define FORE 168
#define NF   8
#define LL   (HIST + FORE)     // 504
#define RR   (BB * HH)         // 5376 rows
#define NTILE (RR / 16)        // 336 WMMA row-tiles

typedef __attribute__((ext_vector_type(16))) _Float16 v16h;
typedef __attribute__((ext_vector_type(8)))  float    v8f;

// -------------------------------------------------------------------------
// k1a: wqk[e] = sum_d Wk[d] * Wq[d,e]  -> Bmat[e,0]; zero padded cols 3..15
__global__ __launch_bounds__(256) void k1a_wqk(const float* __restrict__ Wq,
                                               const float* __restrict__ Wk,
                                               float* __restrict__ Bm) {
    int e = blockIdx.x * 256 + threadIdx.x;
    if (e >= DD) return;
    float s = 0.f;
    for (int d = 0; d < DD; ++d) s += Wk[d] * Wq[d * DD + e];   // coalesced in e
    Bm[e * 16 + 0] = s;
#pragma unroll
    for (int c = 3; c < 16; ++c) Bm[e * 16 + c] = 0.f;
}

// k1b: wvs[d] = Ws[d,:]*Wv -> Bmat[d,1];  bvs[d] = NF*(Ws[d,:]*bv)+bs[d] -> Bmat[d,2]
// one wave per output row d; Ws read once (coalesced), dual accumulators.
__global__ __launch_bounds__(256) void k1b_wvs(const float* __restrict__ Ws,
                                               const float* __restrict__ Wv,
                                               const float* __restrict__ bv,
                                               const float* __restrict__ bs,
                                               float* __restrict__ Bm) {
    int w = threadIdx.x >> 5, lane = threadIdx.x & 31;
    int d = blockIdx.x * 8 + w;                    // grid 64 * 8 waves = 512 rows
    const float* row = Ws + (long)d * DD;
    float a1 = 0.f, a2 = 0.f;
    for (int e = lane; e < DD; e += 32) {
        float r = row[e];
        a1 += r * Wv[e];
        a2 += r * bv[e];
    }
#pragma unroll
    for (int off = 16; off > 0; off >>= 1) {
        a1 += __shfl_down(a1, off, 32);
        a2 += __shfl_down(a2, off, 32);
    }
    if (lane == 0) {
        Bm[d * 16 + 1] = a1;
        Bm[d * 16 + 2] = (float)NF * a2 + bs[d];
    }
}

// k1c: cqk = bq . Wk  (scalar)
__global__ __launch_bounds__(32) void k1c_cqk(const float* __restrict__ bq,
                                              const float* __restrict__ Wk,
                                              float* __restrict__ cqk) {
    int lane = threadIdx.x;
    float s = 0.f;
    for (int d = lane; d < DD; d += 32) s += bq[d] * Wk[d];
#pragma unroll
    for (int off = 16; off > 0; off >>= 1) s += __shfl_down(s, off, 32);
    if (lane == 0) cqk[0] = s;
}

// -------------------------------------------------------------------------
// k2: skinny GEMM T[5376,16] = y_hat[5376,512] x Bmat[512,16] via WMMA f16.
// One wave per 16-row tile. EXEC all-ones (block==32, no divergence) as WMMA
// requires. Lane layouts used here (A/B: contiguous k, not the ISA table order):
//   A (16-bit 16x32): lane grp g=lane/16; element e: K = g*16 + e;
//                     M = lane%16.
//   B (32x16):        element e: K = g*16 + e; N = lane%16.
//   C/D (f32 16x16):  VGPR v: M = g*8 + v; N = lane%16.
__global__ __launch_bounds__(32) void k2_wmma(const float* __restrict__ Y,
                                              const float* __restrict__ Bm,
                                              float* __restrict__ T) {
    const int lane = threadIdx.x;
    const int row  = lane & 15;
    const int grp  = lane >> 4;
    const long r0  = (long)blockIdx.x * 16;
    const float* yrow = Y + (r0 + row) * DD;
    v8f acc = {};
    for (int k0 = 0; k0 < DD; k0 += 32) {
        v16h a, b;
#pragma unroll
        for (int e = 0; e < 8; ++e) {
            a[e]     = (_Float16)yrow[k0 + grp * 16 + e];
            a[e + 8] = (_Float16)yrow[k0 + grp * 16 + 8 + e];
        }
#pragma unroll
        for (int e = 0; e < 16; ++e)
            b[e] = (_Float16)Bm[(k0 + grp * 16 + e) * 16 + row];
        acc = __builtin_amdgcn_wmma_f32_16x16x32_f16(
            /*neg_a=*/false, a, /*neg_b=*/false, b,
            /*c_mod=*/(short)0, acc, /*reuse_a=*/false, /*reuse_b=*/false);
    }
#pragma unroll
    for (int v = 0; v < 8; ++v)
        T[(r0 + grp * 8 + v) * 16 + row] = acc[v];
}

// -------------------------------------------------------------------------
// k3: per (b,h,f): softmax over L of a*x (the qb term cancels in softmax),
// s = E[x] under that softmax; S = sum_f s; logits = (S*a1 + a2)/sqrt(D).
// Block per b: stage full[b] (504x8 f32 = 16KB) in LDS; per-feature col
// min/max computed once per block replaces the per-(h,f) max pass.
__global__ __launch_bounds__(192) void k3_softL(const float* __restrict__ hist,
                                                const float* __restrict__ fore,
                                                const float* __restrict__ T,
                                                const float* __restrict__ cqk,
                                                float* __restrict__ logits) {
    __shared__ float x[LL * NF];
    __shared__ float cmax[NF], cmin[NF];
    const int b = blockIdx.x, tid = threadIdx.x;
    for (int i = tid; i < LL * NF; i += 192)
        x[i] = (i < HIST * NF) ? hist[b * HIST * NF + i]
                               : fore[b * FORE * NF + (i - HIST * NF)];
    __syncthreads();
    if (tid < NF) {
        float mx = -3.4e38f, mn = 3.4e38f;
        for (int l = 0; l < LL; ++l) {
            float v = x[l * NF + tid];
            mx = fmaxf(mx, v); mn = fminf(mn, v);
        }
        cmax[tid] = mx; cmin[tid] = mn;
    }
    __syncthreads();
    if (tid < HH) {
        const float inv_scale = 0.04419417382415922f;   // 1/sqrt(512)
        const int r = b * HH + tid;
        const float a = (T[r * 16 + 0] + cqk[0]) * inv_scale;
        float S = 0.f;
        for (int f = 0; f < NF; ++f) {
            const float m = (a >= 0.f) ? a * cmax[f] : a * cmin[f];
            float se = 0.f, sx = 0.f;
            for (int l = 0; l < LL; ++l) {          // LDS broadcast across wave
                float v = x[l * NF + f];
                float e = __expf(a * v - m);
                se += e; sx += e * v;
            }
            S += sx / se;
        }
        logits[r] = (S * T[r * 16 + 1] + T[r * 16 + 2]) * inv_scale;
    }
}

// -------------------------------------------------------------------------
// k4: per b: P = softmax_h(logits); final[b,d] = sum_h P[h]*y_hat[b,h,d]
__global__ __launch_bounds__(256) void k4_softH(const float* __restrict__ Y,
                                                const float* __restrict__ logits,
                                                float* __restrict__ finalv) {
    __shared__ float sP[HH];
    __shared__ float red[256];
    const int b = blockIdx.x, t = threadIdx.x;
    const float v = (t < HH) ? logits[b * HH + t] : -3.4e38f;
    red[t] = v; __syncthreads();
    for (int s = 128; s > 0; s >>= 1) { if (t < s) red[t] = fmaxf(red[t], red[t + s]); __syncthreads(); }
    const float m = red[0]; __syncthreads();
    const float e = (t < HH) ? __expf(v - m) : 0.f;
    red[t] = e; __syncthreads();
    for (int s = 128; s > 0; s >>= 1) { if (t < s) red[t] += red[t + s]; __syncthreads(); }
    const float inv = 1.f / red[0];
    if (t < HH) sP[t] = e * inv;
    __syncthreads();
    for (int d = t; d < DD; d += 256) {
        float acc = 0.f;
        for (int h = 0; h < HH; ++h)                 // coalesced across threads
            acc += sP[h] * Y[((long)b * HH + h) * DD + d];
        finalv[b * DD + d] = acc;
    }
}

// -------------------------------------------------------------------------
// k5: out[b,h,d] = y_hat[b,h,d] + final[b,d]   (float4 vectorized)
__global__ __launch_bounds__(256) void k5_add(const float4* __restrict__ Y4,
                                              const float4* __restrict__ F4,
                                              float4* __restrict__ O4) {
    const int i = blockIdx.x * 256 + threadIdx.x;   // exactly RR*DD/4 threads
    const int flat = i * 4;
    const int b = flat / (HH * DD);
    const int d = flat % DD;
    const float4 y = Y4[i];
    const float4 f = F4[(b * DD + d) >> 2];
    O4[i] = make_float4(y.x + f.x, y.y + f.y, y.z + f.z, y.w + f.w);
}

// -------------------------------------------------------------------------
extern "C" void kernel_launch(void* const* d_in, const int* in_sizes, int n_in,
                              void* d_out, int out_size, void* d_ws, size_t ws_size,
                              hipStream_t stream) {
    const float* y_hat = (const float*)d_in[0];
    const float* whist = (const float*)d_in[1];
    const float* wfore = (const float*)d_in[2];
    const float* Wq    = (const float*)d_in[3];
    const float* bq    = (const float*)d_in[4];
    const float* Wk    = (const float*)d_in[5];
    // d_in[6] = bk: cancels inside the L-softmax (constant in L) -> unused.
    const float* Wv    = (const float*)d_in[7];
    const float* bv    = (const float*)d_in[8];
    const float* Ws    = (const float*)d_in[9];
    const float* bs    = (const float*)d_in[10];
    float* out = (float*)d_out;

    // workspace carve (all f32): ~464 KB total
    float* Bm     = (float*)d_ws;          // 512*16
    float* cqk    = Bm  + DD * 16;         // 1 (+15 pad)
    float* T      = cqk + 16;              // 5376*16
    float* logitsv= T   + RR * 16;         // 5376
    float* finalv = logitsv + RR;          // 32*512

    k1a_wqk<<<2, 256, 0, stream>>>(Wq, Wk, Bm);
    k1b_wvs<<<DD / 8, 256, 0, stream>>>(Ws, Wv, bv, bs, Bm);
    k1c_cqk<<<1, 32, 0, stream>>>(bq, Wk, cqk);
    k2_wmma<<<NTILE, 32, 0, stream>>>(y_hat, Bm, T);
    k3_softL<<<BB, 192, 0, stream>>>(whist, wfore, T, cqk, logitsv);
    k4_softH<<<BB, 256, 0, stream>>>(y_hat, logitsv, finalv);
    k5_add<<<(RR * DD / 4) / 256, 256, 0, stream>>>(
        (const float4*)y_hat, (const float4*)finalv, (float4*)out);
}
